// SpikingLinearLayer_30159260352832
// MI455X (gfx1250) — matrix operand order repaired and checked by reading, not compiled, not run on hardware
//
#include <hip/hip_runtime.h>
#include <stdint.h>

typedef __attribute__((ext_vector_type(16))) _Float16 v16h;
typedef __attribute__((ext_vector_type(8)))  _Float16 v8h;
typedef __attribute__((ext_vector_type(8)))  float    v8f;

// Scheduling-group hints (force DS-read burst, then WMMA burst, so the
// waitcnt pass can emit partial s_wait_dscnt instead of full stalls).
#if __has_builtin(__builtin_amdgcn_sched_group_barrier)
#define SGB(mask, cnt, id) __builtin_amdgcn_sched_group_barrier((mask), (cnt), (id))
#else
#define SGB(mask, cnt, id)
#endif

// Problem dims (fixed by reference setup_inputs)
constexpr int T    = 20;
constexpr int Bdim = 1024;
constexpr int IN   = 2048;
constexpr int OUT  = 2048;

// LIF constants
constexpr float ALPHA_M = 1.0f - 1.0f / 20.0f;  // 0.95
constexpr float ALPHA_S = 1.0f - 1.0f / 5.0f;   // 0.80
constexpr float KAPPA   = 1.0f / 20.0f;         // DT/TAU_M
constexpr float THRESH  = 1.0f;

// Tiling
constexpr int BM = 64;    // batch rows per block
constexpr int BN = 128;   // out cols per block
constexpr int KC = 64;    // K-chunk staged in LDS (2 WMMA k-steps of 32)
constexpr int KITERS = IN / KC;  // 32

// ---------------------------------------------------------------------------
// f32 -> f16 conversion (x spikes are {0,1}: exact; weights: ~1e-3 rel err)
// ---------------------------------------------------------------------------
__global__ __launch_bounds__(256) void cvt_f32_to_f16(
    const float* __restrict__ src, _Float16* __restrict__ dst, long n) {
  long i = ((long)blockIdx.x * 256 + threadIdx.x) * 8;
  if (i >= n) return;
  v8h h;
#pragma unroll
  for (int j = 0; j < 8; ++j) h[j] = (_Float16)src[i + j];
  *(v8h*)(dst + i) = h;  // global_store_b128
}

// ---------------------------------------------------------------------------
// Issue one K-chunk's worth of async global->LDS b128 DMAs for this thread:
// 2 x 16B of the X tile (64 x KC) and 4 x 16B of the W tile (128 x KC).
// LDS layout: row-major [row][KC] halfs, lin = row*8 + c  <->  bytes lin*16.
// ---------------------------------------------------------------------------
__device__ __forceinline__ void issue_chunk(
    const _Float16* __restrict__ xh, const _Float16* __restrict__ wh,
    int t, int kc, unsigned ldsX, unsigned ldsW, int b0, int n0, int tid) {
  const unsigned long long xg = (unsigned long long)(uintptr_t)(
      xh + ((size_t)t * Bdim + b0) * IN + (size_t)kc * KC);
  const unsigned long long wg = (unsigned long long)(uintptr_t)(
      wh + (size_t)n0 * IN + (size_t)kc * KC);
#pragma unroll
  for (int j = 0; j < 2; ++j) {
    const int lin = j * 256 + tid;
    const int r = lin >> 3, c = lin & 7;
    const unsigned long long ga = xg + ((size_t)r * IN + (size_t)c * 8) * 2ull;
    const unsigned la = ldsX + (unsigned)lin * 16u;
    asm volatile("global_load_async_to_lds_b128 %0, %1, off"
                 :: "v"(la), "v"(ga) : "memory");
  }
#pragma unroll
  for (int j = 0; j < 4; ++j) {
    const int lin = j * 256 + tid;
    const int r = lin >> 3, c = lin & 7;
    const unsigned long long ga = wg + ((size_t)r * IN + (size_t)c * 8) * 2ull;
    const unsigned la = ldsW + (unsigned)lin * 16u;
    asm volatile("global_load_async_to_lds_b128 %0, %1, off"
                 :: "v"(la), "v"(ga) : "memory");
  }
}

// ---------------------------------------------------------------------------
// Fused SNN: per-block 64x128 output tile, t-outer loop, K streamed through
// double-buffered LDS via async global->LDS DMA (ASYNCcnt), cur accumulated
// with v_wmma_f32_16x16x32_f16, LIF state (V, I) in registers across T.
// ---------------------------------------------------------------------------
__global__ __launch_bounds__(256) void snn_fused(
    const _Float16* __restrict__ xh,   // [T, B, IN]  f16
    const _Float16* __restrict__ wh,   // [OUT, IN]   f16
    float* __restrict__ out)           // [T, B, OUT] f32 spikes
{
  __shared__ _Float16 sX[2][BM * KC];  // 2 x 8 KB
  __shared__ _Float16 sW[2][BN * KC];  // 2 x 16 KB

  const int tid   = threadIdx.x;
  const int wave  = tid >> 5;      // 0..7
  const int lane  = tid & 31;
  const int lrow  = lane & 15;     // row/col within 16
  const int lhalf = lane >> 4;     // 0/1: which K/M half this lane holds
  const int wm    = wave & 3;      // 16-row band within BM
  const int wn    = wave >> 2;     // 64-col band within BN

  const int n0 = blockIdx.x * BN;  // out-col base
  const int b0 = blockIdx.y * BM;  // batch-row base

  // LDS byte addresses (low 32 bits of the generic shared pointers)
  const unsigned ldsXb[2] = {(unsigned)(uintptr_t)&sX[0][0],
                             (unsigned)(uintptr_t)&sX[1][0]};
  const unsigned ldsWb[2] = {(unsigned)(uintptr_t)&sW[0][0],
                             (unsigned)(uintptr_t)&sW[1][0]};

  // LIF state per owned C element (C layout: M = j + 8*lhalf, N = lrow)
  float V[4][8], I[4][8];
#pragma unroll
  for (int nt = 0; nt < 4; ++nt)
#pragma unroll
    for (int j = 0; j < 8; ++j) { V[nt][j] = 0.0f; I[nt][j] = 0.0f; }

  v8f acc[4];
  int buf = 0;

  // Prologue: DMA chunk (t=0, kc=0) into buffer 0.
  issue_chunk(xh, wh, 0, 0, ldsXb[0], ldsWb[0], b0, n0, tid);

#pragma unroll 1
  for (int t = 0; t < T; ++t) {
#pragma unroll
    for (int nt = 0; nt < 4; ++nt)
#pragma unroll
      for (int j = 0; j < 8; ++j) acc[nt][j] = 0.0f;

#pragma unroll 1
    for (int kc = 0; kc < KITERS; ++kc) {
      // Own async DMAs for the current buffer complete...
      asm volatile("s_wait_asynccnt 0x0" ::: "memory");
      // ...then everyone's are, and all waves finished the previous chunk's
      // compute (which read the other buffer) -> safe to refill it now.
      __syncthreads();

      const int nkc = (kc + 1 < KITERS) ? kc + 1 : 0;
      const int ntm = (kc + 1 < KITERS) ? t : t + 1;
      if (ntm < T)  // uniform across block
        issue_chunk(xh, wh, ntm, nkc, ldsXb[buf ^ 1], ldsWb[buf ^ 1], b0, n0, tid);

      // ---- 2 k-steps of 32; force [10x DS-read][4x WMMA] per k-step so
      //      fragment loads stay live in distinct regs -> partial waits ----
      const _Float16* xs = sX[buf];
      const _Float16* ws = sW[buf];
#pragma unroll
      for (int kk = 0; kk < KC / 32; ++kk) {
        // A fragment 16x32: lane(lrow,lhalf) holds K = 16*lhalf+[0,16)
        // of row wm*16+lrow
        const _Float16* p = xs + (wm * 16 + lrow) * KC + kk * 32;
        // B fragments 32x16: lane holds K = 16*lhalf+[0,16) of column lrow;
        // N-tiles are 16 rows (= 16*KC halfs) apart in the W tile.
        const _Float16* q = ws + (wn * 64 + lrow) * KC + kk * 32 + lhalf * 16;

        v16h a, bf0, bf1, bf2, bf3;
        *(v8h*)&a         = *(const v8h*)(p + lhalf * 16);
        *((v8h*)&a + 1)   = *(const v8h*)(p + lhalf * 16 + 8);
        *(v8h*)&bf0       = *(const v8h*)(q);
        *((v8h*)&bf0 + 1) = *(const v8h*)(q + 8);
        *(v8h*)&bf1       = *(const v8h*)(q + 16 * KC);
        *((v8h*)&bf1 + 1) = *(const v8h*)(q + 16 * KC + 8);
        *(v8h*)&bf2       = *(const v8h*)(q + 32 * KC);
        *((v8h*)&bf2 + 1) = *(const v8h*)(q + 32 * KC + 8);
        *(v8h*)&bf3       = *(const v8h*)(q + 48 * KC);
        *((v8h*)&bf3 + 1) = *(const v8h*)(q + 48 * KC + 8);

        acc[0] = __builtin_amdgcn_wmma_f32_16x16x32_f16(
            false, a, false, bf0, (short)0, acc[0], false, false);
        acc[1] = __builtin_amdgcn_wmma_f32_16x16x32_f16(
            false, a, false, bf1, (short)0, acc[1], false, false);
        acc[2] = __builtin_amdgcn_wmma_f32_16x16x32_f16(
            false, a, false, bf2, (short)0, acc[2], false, false);
        acc[3] = __builtin_amdgcn_wmma_f32_16x16x32_f16(
            false, a, false, bf3, (short)0, acc[3], false, false);

        SGB(0x100, 10, 0);  // 10 DS reads first...
        SGB(0x008,  4, 0);  // ...then 4 WMMA (matrix ops)
      }
      buf ^= 1;
    }  // kc

    // ---- LIF update + spike store (elementwise on accumulator layout) ----
#pragma unroll
    for (int nt = 0; nt < 4; ++nt) {
      const int ng = n0 + wn * 64 + nt * 16 + lrow;
#pragma unroll
      for (int j = 0; j < 8; ++j) {
        const float cur = acc[nt][j];
        const float In = ALPHA_S * I[nt][j] + cur;
        const float Vn = ALPHA_M * V[nt][j] + KAPPA * In;
        const bool fire = (Vn >= THRESH);
        I[nt][j] = In;
        V[nt][j] = fire ? 0.0f : Vn;
        const int mg = b0 + wm * 16 + j + 8 * lhalf;
        out[(size_t)t * Bdim * OUT + (size_t)mg * OUT + ng] = fire ? 1.0f : 0.0f;
      }
    }
  }  // t
}

// ---------------------------------------------------------------------------
extern "C" void kernel_launch(void* const* d_in, const int* in_sizes, int n_in,
                              void* d_out, int out_size, void* d_ws, size_t ws_size,
                              hipStream_t stream) {
  const float* x = (const float*)d_in[0];   // [T,B,IN] spikes (f32 0/1)
  const float* w = (const float*)d_in[1];   // [OUT,IN] f32
  float* out = (float*)d_out;               // [T,B,OUT] f32

  _Float16* wh = (_Float16*)d_ws;           // 8 MB
  _Float16* xh = wh + (size_t)OUT * IN;     // 84 MB

  const long nw = (long)OUT * IN;           // 4 Mi elements
  const long nx = (long)T * Bdim * IN;      // 40 Mi elements

  cvt_f32_to_f16<<<(int)(nw / (256 * 8)), 256, 0, stream>>>(w, wh, nw);
  cvt_f32_to_f16<<<(int)(nx / (256 * 8)), 256, 0, stream>>>(x, xh, nx);

  dim3 grid(OUT / BN, Bdim / BM);           // 16 x 16 = 256 blocks
  snn_fused<<<grid, 256, 0, stream>>>(xh, wh, out);
}
